// Attention_4904852652579
// MI455X (gfx1250) — matrix operand order repaired and checked by reading, not compiled, not run on hardware
//
#include <hip/hip_runtime.h>
#include <hip/hip_bf16.h>

#define DIM   1024
#define HEADS 16
#define BATCH 2
#define SEQ   2048
#define DH    64
#define ROWS  (BATCH * SEQ)      // 4096
#define LDSTR 72                 // padded LDS row stride (elements): 36 dwords -> conflict-free

typedef __attribute__((ext_vector_type(16))) __bf16 v16bf;
typedef __attribute__((ext_vector_type(8)))  float  v8f;
typedef __attribute__((ext_vector_type(4)))  int    v4i;

union Frag { v16bf v; uint4 q[2]; };

__device__ __forceinline__ __bf16 f2bf(float f) {
    union { float f; unsigned u; } x; x.f = f;
    unsigned r = x.u + 0x7FFFu + ((x.u >> 16) & 1u);   // round-to-nearest-even
    unsigned short h = (unsigned short)(r >> 16);
    return __builtin_bit_cast(__bf16, h);
}

// ---- async global->LDS copy (gfx1250), with portable fallback ----
#if defined(__has_builtin)
#if __has_builtin(__builtin_amdgcn_global_load_async_to_lds_b128)
#define HAVE_ASYNC_LDS 1
#endif
#if __has_builtin(__builtin_amdgcn_s_wait_asynccnt)
#define HAVE_WAIT_ASYNC 1
#endif
#endif

__device__ __forceinline__ void async_cp16(const __bf16* g, __bf16* l) {
#ifdef HAVE_ASYNC_LDS
    __builtin_amdgcn_global_load_async_to_lds_b128(
        (__attribute__((address_space(1))) v4i*)g,
        (__attribute__((address_space(3))) v4i*)l, 0, 0);
#else
    *(uint4*)l = *(const uint4*)g;                      // sync fallback
#endif
}

#ifdef HAVE_WAIT_ASYNC
#define WAIT_ASYNC(n) __builtin_amdgcn_s_wait_asynccnt(n)
#else
#define WAIT_ASYNC(n) asm volatile("s_wait_asynccnt " #n ::: "memory")
#endif

// ---------------- prep kernels ----------------
__global__ void cvt_f32_to_bf16(const float* __restrict__ in, __bf16* __restrict__ out, int n) {
    int i = blockIdx.x * 256 + threadIdx.x;
    if (i < n) out[i] = f2bf(in[i]);
}

// in[K][N] row-major -> out[N][K] (bf16), so GEMM B reads are K-contiguous
__global__ void transpose_to_bf16(const float* __restrict__ in, __bf16* __restrict__ out, int K, int N) {
    int i = blockIdx.x * 256 + threadIdx.x;
    if (i < K * N) {
        int k = i / N, n = i % N;
        out[(size_t)n * K + k] = f2bf(in[i]);
    }
}

// ---------------- WMMA GEMM with async-LDS double buffering ----------------
// C[M,N] = A[M,K](bf16 row-major) * Bt[N,K](bf16 pre-transposed)
// Block: 256 thr = 8 waves as 4(M) x 2(N); wave tile 32x64 = 2x4 WMMA frags.
// K chunked by 64; 128x64 tiles of A and B staged in LDS (double buffered).
// mode 0: write f32 to Cf[M][N]
// mode 1: scatter qkv: q[b,h,n,dh], k[b,h,n,dh], vT[b,h,dh,n] (bf16)

// coalesced 128x64 bf16 tile copy: 8 lanes cover one 128B line
__device__ __forceinline__ void stage128x64(const __bf16* __restrict__ g, int ldg,
                                            __bf16* l, int tid) {
    const int r = tid >> 3;            // 0..31
    const int c = (tid & 7) * 8;       // 0..56
#pragma unroll
    for (int i = 0; i < 4; i++)
        async_cp16(g + (size_t)(r + 32 * i) * ldg + c,
                   l + (r + 32 * i) * LDSTR + c);
}

__global__ __launch_bounds__(256)
void gemm_wmma(const __bf16* __restrict__ A, const __bf16* __restrict__ Bt,
               float* __restrict__ Cf,
               __bf16* __restrict__ qb, __bf16* __restrict__ kbf, __bf16* __restrict__ vT,
               int M, int N, int K, int mode)
{
    __shared__ alignas(16) __bf16 sA[2][128 * LDSTR];
    __shared__ alignas(16) __bf16 sB[2][128 * LDSTR];

    const int tid  = threadIdx.x;
    const int lane = tid & 31;
    const int wave = tid >> 5;
    const int wm = wave >> 1;            // 0..3
    const int wn = wave & 1;             // 0..1
    const int lg = lane >> 4;            // lane half
    const int ll = lane & 15;

    const __bf16* Ab  = A  + (size_t)(blockIdx.y * 128) * K;
    const __bf16* Btb = Bt + (size_t)(blockIdx.x * 128) * K;

    v8f acc[2][4];
#pragma unroll
    for (int i = 0; i < 2; i++)
#pragma unroll
        for (int j = 0; j < 4; j++)
            acc[i][j] = (v8f){0.f,0.f,0.f,0.f,0.f,0.f,0.f,0.f};

    const int nk = K >> 6;               // K chunks of 64

    // prologue: stage chunk 0 into buffer 0
    stage128x64(Ab, K, sA[0], tid);
    stage128x64(Btb, K, sB[0], tid);

    for (int kc = 0; kc < nk; kc++) {
        const int buf = kc & 1;
        if (kc + 1 < nk) {
            stage128x64(Ab  + (kc + 1) * 64, K, sA[buf ^ 1], tid);
            stage128x64(Btb + (kc + 1) * 64, K, sB[buf ^ 1], tid);
            WAIT_ASYNC(8);               // chunk kc landed; chunk kc+1 in flight
        } else {
            WAIT_ASYNC(0);
        }
        __syncthreads();                 // chunk kc visible to whole block

#pragma unroll
        for (int ks = 0; ks < 64; ks += 32) {
            Frag a[2], b[4];
#pragma unroll
            for (int mt = 0; mt < 2; mt++) {
                const __bf16* p = &sA[buf][(wm * 32 + mt * 16 + ll) * LDSTR + ks + lg * 8];
                a[mt].q[0] = *(const uint4*)p;
                a[mt].q[1] = *(const uint4*)(p + 16);
            }
#pragma unroll
            for (int nt = 0; nt < 4; nt++) {
                const __bf16* p = &sB[buf][(wn * 64 + nt * 16 + ll) * LDSTR + ks + lg * 8];
                b[nt].q[0] = *(const uint4*)p;
                b[nt].q[1] = *(const uint4*)(p + 16);
            }
#pragma unroll
            for (int mt = 0; mt < 2; mt++)
#pragma unroll
                for (int nt = 0; nt < 4; nt++)
                    acc[mt][nt] = __builtin_amdgcn_wmma_f32_16x16x32_bf16(
                        false, a[mt].v, false, b[nt].v, (short)0, acc[mt][nt], false, false);
        }
        __syncthreads();                 // all waves done reading buf before it is re-staged
    }

    // epilogue: C layout — row m = lg*8 + r, col = ll
    const int m_base = blockIdx.y * 128 + wm * 32;
    const int n_base = blockIdx.x * 128 + wn * 64;
#pragma unroll
    for (int mt = 0; mt < 2; mt++)
#pragma unroll
        for (int nt = 0; nt < 4; nt++)
#pragma unroll
            for (int r = 0; r < 8; r++) {
                int m = m_base + mt * 16 + lg * 8 + r;
                int c = n_base + nt * 16 + ll;
                float v = acc[mt][nt][r];
                if (mode == 0) {
                    Cf[(size_t)m * N + c] = v;
                } else {
                    int bb = m >> 11, tok = m & (SEQ - 1);
                    int which = c >> 10, d = c & (DIM - 1);
                    int h = d >> 6, dh = d & (DH - 1);
                    int bh = bb * HEADS + h;
                    if (which == 0)      qb [((size_t)bh * SEQ + tok) * DH + dh] = f2bf(v);
                    else if (which == 1) kbf[((size_t)bh * SEQ + tok) * DH + dh] = f2bf(v);
                    else                 vT [((size_t)bh * DH + dh) * SEQ + tok] = f2bf(v);
                }
            }
}

// ---------------- flash attention ----------------
// One wave per 16 query rows of one (b,h). Block = 4 waves (independent).
__global__ __launch_bounds__(128)
void flash_attn(const __bf16* __restrict__ qb, const __bf16* __restrict__ kbf,
                const __bf16* __restrict__ vT, __bf16* __restrict__ attn)
{
    __shared__ alignas(16) __bf16 lds_p[4][16][32];   // per-wave P tile (16 q rows x 32 keys)

    const int lane = threadIdx.x & 31;
    const int wave = threadIdx.x >> 5;
    const int gw = blockIdx.x * 4 + wave;           // 0..4095
    const int qt = gw & 127;                        // query tile
    const int h  = (gw >> 7) & (HEADS - 1);
    const int b  = gw >> 11;
    const int bh = b * HEADS + h;
    const int q0 = qt * 16;
    const int lg = lane >> 4, ll = lane & 15;

    const float scale = 0.125f;                     // Dh^-0.5
    const float slope = exp2f(-0.5f * (float)(h + 1));

    // Q A-frags (row = q0+ll, Dh split into two K=32 chunks)
    Frag aq[2];
#pragma unroll
    for (int dc = 0; dc < 2; dc++) {
        const __bf16* p = qb + ((size_t)bh * SEQ + q0 + ll) * DH + dc * 32 + lg * 8;
        aq[dc].q[0] = *(const uint4*)p;
        aq[dc].q[1] = *(const uint4*)(p + 16);
    }

    v8f o[4];
#pragma unroll
    for (int dt = 0; dt < 4; dt++) o[dt] = (v8f){0.f,0.f,0.f,0.f,0.f,0.f,0.f,0.f};
    float row_m[8], row_l[8];
#pragma unroll
    for (int r = 0; r < 8; r++) { row_m[r] = -1e30f; row_l[r] = 0.f; }

    const int nblocks = (q0 + 47) >> 5;             // covers keys 0..q0+15
    for (int blk = 0; blk < nblocks; blk++) {
        const int kp = blk * 32;

        // S = Q * K^T : two 16x16 key tiles
        v8f s[2];
        s[0] = (v8f){0.f,0.f,0.f,0.f,0.f,0.f,0.f,0.f};
        s[1] = (v8f){0.f,0.f,0.f,0.f,0.f,0.f,0.f,0.f};
#pragma unroll
        for (int jt = 0; jt < 2; jt++)
#pragma unroll
            for (int dc = 0; dc < 2; dc++) {
                Frag bk;
                const __bf16* p = kbf + ((size_t)bh * SEQ + kp + jt * 16 + ll) * DH + dc * 32 + lg * 8;
                bk.q[0] = *(const uint4*)p;
                bk.q[1] = *(const uint4*)(p + 16);
                s[jt] = __builtin_amdgcn_wmma_f32_16x16x32_bf16(
                    false, aq[dc].v, false, bk.v, (short)0, s[jt], false, false);
            }

        // scale + ALiBi + causal mask + online softmax; P -> LDS (bf16, A-layout bounce)
        const int j0 = kp + ll, j1 = kp + 16 + ll;
#pragma unroll
        for (int r = 0; r < 8; r++) {
            const int mg = q0 + lg * 8 + r;
            float v0 = s[0][r] * scale + slope * (float)j0;
            float v1 = s[1][r] * scale + slope * (float)j1;
            if (j0 > mg) v0 = -1e30f;
            if (j1 > mg) v1 = -1e30f;
            float mx = fmaxf(v0, v1);
            for (int off = 1; off < 16; off <<= 1) mx = fmaxf(mx, __shfl_xor(mx, off, 32));
            const float nm = fmaxf(row_m[r], mx);
            const float alpha = __expf(row_m[r] - nm);
            row_m[r] = nm;
            const float p0 = __expf(v0 - nm);
            const float p1 = __expf(v1 - nm);
            float ps = p0 + p1;
            for (int off = 1; off < 16; off <<= 1) ps += __shfl_xor(ps, off, 32);
            row_l[r] = row_l[r] * alpha + ps;
#pragma unroll
            for (int dt = 0; dt < 4; dt++) o[dt][r] *= alpha;
            const int ml = lg * 8 + r;
            lds_p[wave][ml][ll]      = f2bf(p0);
            lds_p[wave][ml][ll + 16] = f2bf(p1);
        }
        asm volatile("s_wait_dscnt 0" ::: "memory");

        // P A-frag from LDS
        Frag ap;
        const __bf16* pp = &lds_p[wave][ll][lg * 8];
        ap.q[0] = *(const uint4*)pp;
        ap.q[1] = *(const uint4*)(pp + 16);

        // O += P * V  (V pre-transposed: vT[b,h,dh,n] -> K-contiguous B frags)
#pragma unroll
        for (int dt = 0; dt < 4; dt++) {
            Frag bv;
            const __bf16* p = vT + ((size_t)bh * DH + dt * 16 + ll) * SEQ + kp + lg * 8;
            bv.q[0] = *(const uint4*)p;
            bv.q[1] = *(const uint4*)(p + 16);
            o[dt] = __builtin_amdgcn_wmma_f32_16x16x32_bf16(
                false, ap.v, false, bv.v, (short)0, o[dt], false, false);
        }
    }

    // normalize and write attn[b, tok, h*64 + dh] as bf16 ([ROWS][DIM] for final GEMM)
#pragma unroll
    for (int r = 0; r < 8; r++) {
        const float inv = 1.0f / row_l[r];
        const int tok = q0 + lg * 8 + r;
#pragma unroll
        for (int dt = 0; dt < 4; dt++) {
            float v = o[dt][r] * inv;
            attn[((size_t)b * SEQ + tok) * DIM + h * DH + dt * 16 + ll] = f2bf(v);
        }
    }
}

// ---------------- launch ----------------
extern "C" void kernel_launch(void* const* d_in, const int* in_sizes, int n_in,
                              void* d_out, int out_size, void* d_ws, size_t ws_size,
                              hipStream_t stream) {
    const float* x      = (const float*)d_in[0];   // [2,2048,1024]
    const float* w_qkv  = (const float*)d_in[1];   // [1024,3072]
    const float* w_proj = (const float*)d_in[2];   // [1024,1024]
    float* out = (float*)d_out;                    // [2,2048,1024] f32

    char* ws = (char*)d_ws;
    size_t off = 0;
    auto carve = [&](size_t bytes) { void* p = ws + off; off += (bytes + 255) & ~(size_t)255; return p; };

    __bf16* xb     = (__bf16*)carve((size_t)ROWS * DIM * 2);
    __bf16* wqkvT  = (__bf16*)carve((size_t)3 * DIM * DIM * 2);
    __bf16* wprojT = (__bf16*)carve((size_t)DIM * DIM * 2);
    __bf16* qb     = (__bf16*)carve((size_t)BATCH * HEADS * SEQ * DH * 2);
    __bf16* kbf    = (__bf16*)carve((size_t)BATCH * HEADS * SEQ * DH * 2);
    __bf16* vT     = (__bf16*)carve((size_t)BATCH * HEADS * SEQ * DH * 2);
    __bf16* attn   = (__bf16*)carve((size_t)ROWS * DIM * 2);

    { int n = ROWS * DIM;
      cvt_f32_to_bf16<<<(n + 255) / 256, 256, 0, stream>>>(x, xb, n); }
    { int n = DIM * 3 * DIM;
      transpose_to_bf16<<<(n + 255) / 256, 256, 0, stream>>>(w_qkv, wqkvT, DIM, 3 * DIM); }
    { int n = DIM * DIM;
      transpose_to_bf16<<<(n + 255) / 256, 256, 0, stream>>>(w_proj, wprojT, DIM, DIM); }

    // qkv = x @ w_qkv  (scatter into q/k/vT)
    { dim3 g(3 * DIM / 128, ROWS / 128);
      gemm_wmma<<<g, 256, 0, stream>>>(xb, wqkvT, nullptr, qb, kbf, vT, ROWS, 3 * DIM, DIM, 1); }

    // attention
    { int total_waves = BATCH * HEADS * (SEQ / 16);   // 4096
      flash_attn<<<total_waves / 4, 128, 0, stream>>>(qb, kbf, vT, attn); }

    // out = attn @ w_proj
    { dim3 g(DIM / 128, ROWS / 128);
      gemm_wmma<<<g, 256, 0, stream>>>(attn, wprojT, out, nullptr, nullptr, nullptr, ROWS, DIM, DIM, 0); }
}
